// MolGINE_18588618457330
// MI455X (gfx1250) — matrix operand order repaired and checked by reading, not compiled, not run on hardware
//
#include <hip/hip_runtime.h>

typedef __attribute__((ext_vector_type(16))) _Float16 v16h;
typedef __attribute__((ext_vector_type(8)))  _Float16 v8h;
typedef __attribute__((ext_vector_type(4)))  _Float16 v4h;
typedef __attribute__((ext_vector_type(8)))  float    v8f;
typedef __attribute__((ext_vector_type(4)))  float    v4f;

#define NN   100000   // nodes  (multiple of 32)
#define NE   200000   // edges
#define NG   4096     // graphs (multiple of 32)
#define DIM  256
#define HID  512
#define ODIM 768
#define NF   9
#define EF   3
#define NVOC 119
#define EVOC 22
#define NLAY 4

// ---------------------------------------------------------------- encoders
__global__ void k_encode_nodes(const int* __restrict__ xcat,
                               const float* __restrict__ emb,
                               float* __restrict__ x) {
  int n = blockIdx.x, d = threadIdx.x;           // 100000 blocks x 256
  float acc = 0.f;
#pragma unroll
  for (int f = 0; f < NF; ++f) {
    int idx = xcat[n * NF + f];
    acc += emb[((long)f * NVOC + idx) * DIM + d];
  }
  x[(long)n * DIM + d] = acc;
}

__global__ void k_encode_edges(const int* __restrict__ eattr,
                               const float* __restrict__ emb,
                               _Float16* __restrict__ e16) {
  int e = blockIdx.x, d = threadIdx.x;           // 200000 blocks x 256
  float acc = 0.f;
#pragma unroll
  for (int f = 0; f < EF; ++f) {
    int idx = eattr[e * EF + f];
    acc += emb[((long)f * EVOC + idx) * DIM + d];
  }
  e16[(long)e * DIM + d] = (_Float16)acc;
}

// ---------------------------------------------------------------- utility
__global__ void k_copy_f32(const float* __restrict__ a, float* __restrict__ b, long n) {
  long t = (long)blockIdx.x * 256 + threadIdx.x;
  if (t < n) b[t] = a[t];
}
__global__ void k_zero_f32(float* __restrict__ a, long n) {
  long t = (long)blockIdx.x * 256 + threadIdx.x;
  if (t < n) a[t] = 0.f;
}
__global__ void k_f32_to_f16(const float* __restrict__ a, _Float16* __restrict__ b, long n) {
  long t = (long)blockIdx.x * 256 + threadIdx.x;
  if (t < n) b[t] = (_Float16)a[t];
}

// Repack row-major W[K][N] (fp32) into WMMA B-fragment order (f16):
// flat = ((nTile*(K/32) + kTile)*32 + lane)*16 + j
//   n = nTile*16 + (lane&15);  k = kTile*32 + (lane>>4)*16 + j
__global__ void k_repack_w(const float* __restrict__ W, _Float16* __restrict__ out,
                           int K, int N) {
  long t = (long)blockIdx.x * 256 + threadIdx.x;
  long total = (long)K * N;
  if (t >= total) return;
  int j      = (int)(t & 15);
  int lane   = (int)((t >> 4) & 31);
  long rest  = t >> 9;
  int kTiles = K >> 5;
  int kTile  = (int)(rest % kTiles);
  int nTile  = (int)(rest / kTiles);
  int n = nTile * 16 + (lane & 15);
  int k = kTile * 32 + (lane >> 4) * 16 + j;
  out[t] = (_Float16)W[(long)k * N + n];
}

// ---------------------------------------------------------------- scatter
// h[dst] += relu(x[src] + e) ; 64 threads/edge, 4 dims each
__global__ void k_edge_scatter(const float* __restrict__ x,
                               const _Float16* __restrict__ e16,
                               const int* __restrict__ ei,
                               float* __restrict__ h) {
  long t = (long)blockIdx.x * 256 + threadIdx.x;
  if (t >= (long)NE * 64) return;
  int eid = (int)(t >> 6);
  int db  = (int)(t & 63) << 2;
  int src = ei[eid];
  int dst = ei[NE + eid];
  v4f xv = *(const v4f*)(x + (long)src * DIM + db);
  v4h ev = *(const v4h*)(e16 + (long)eid * DIM + db);
  float* hp = h + (long)dst * DIM + db;
#pragma unroll
  for (int i = 0; i < 4; ++i) {
    float m = xv[i] + (float)ev[i];
    m = m > 0.f ? m : 0.f;
    unsafeAtomicAdd(hp + i, m);       // hardware global_atomic_add_f32
  }
}

__global__ void k_pool(const float* __restrict__ x, const int* __restrict__ batch,
                       float* __restrict__ g) {
  long t = (long)blockIdx.x * 256 + threadIdx.x;
  if (t >= (long)NN * DIM) return;
  int n = (int)(t >> 8), d = (int)(t & 255);
  unsafeAtomicAdd(g + (long)batch[n] * DIM + d, x[t]);
}

// ---------------------------------------------------------------- WMMA GEMM
// C[M,N] = op(A[M,K] @ Bpk + bias); A f16 row-major, Bpk pre-fragmented f16.
// Block = 256 thr = 8 waves; block covers 256 rows x 64 cols.
// Each wave owns 32 rows x 64 cols: 2 row tiles x 4 col tiles = 8 accums,
// so each loaded B fragment feeds 2 WMMAs (halves B VMEM per FLOP).
// Requires M % 32 == 0 (holds: 100000, 4096).
template <bool RELU, bool OUT16>
__global__ void k_gemm(const _Float16* __restrict__ A,
                       const _Float16* __restrict__ Bpk,
                       const float* __restrict__ bias,
                       float* __restrict__ Cf, _Float16* __restrict__ Ch,
                       int M, int N, int K) {
  int wave = threadIdx.x >> 5;
  int lane = threadIdx.x & 31;
  int rowBase = blockIdx.x * 256 + wave * 32;
  if (rowBase >= M) return;                       // uniform per wave; M%32==0
  int colBase = blockIdx.y * 64;
  int laneLo = lane & 15, laneHi = lane >> 4;
  const _Float16* arow0 = A + (long)(rowBase + laneLo) * K;
  const _Float16* arow1 = A + (long)(rowBase + 16 + laneLo) * K;
  int kTiles = K >> 5;

  v8f acc0[4] = {v8f{}, v8f{}, v8f{}, v8f{}};
  v8f acc1[4] = {v8f{}, v8f{}, v8f{}, v8f{}};

  for (int kt = 0; kt < kTiles; ++kt) {
    int kb = kt << 5;
    // A fragments: contiguous K (lanes 0-15: K 0..15; 16-31: K 16..31), matches Bpk
    union { v16h v; v8h h[2]; } a0, a1;
    a0.h[0] = *(const v8h*)(arow0 + kb + laneHi * 16);
    a0.h[1] = *(const v8h*)(arow0 + kb + laneHi * 16 + 8);
    a1.h[0] = *(const v8h*)(arow1 + kb + laneHi * 16);
    a1.h[1] = *(const v8h*)(arow1 + kb + laneHi * 16 + 8);
    if (kb + 32 < K) {
      __builtin_prefetch(arow0 + kb + 32, 0, 1);  // global_prefetch_b8
      __builtin_prefetch(arow1 + kb + 32, 0, 1);
    }
#pragma unroll
    for (int t = 0; t < 4; ++t) {
      int nTile = (colBase >> 4) + t;
      const v8h* bp =
          (const v8h*)(Bpk + (((long)nTile * kTiles + kt) * 32 + lane) * 16);
      union { v16h v; v8h h[2]; } b;
      b.h[0] = bp[0];
      b.h[1] = bp[1];
      acc0[t] = __builtin_amdgcn_wmma_f32_16x16x32_f16(
          false, a0.v, false, b.v, (short)0, acc0[t], false, false);
      acc1[t] = __builtin_amdgcn_wmma_f32_16x16x32_f16(
          false, a1.v, false, b.v, (short)0, acc1[t], false, false);
    }
  }

#pragma unroll
  for (int t = 0; t < 4; ++t) {
    int n = colBase + t * 16 + laneLo;
    float bv = bias ? bias[n] : 0.f;
#pragma unroll
    for (int r = 0; r < 8; ++r) {                 // D layout: M = r + 8*laneHi
      int m0 = rowBase + laneHi * 8 + r;
      int m1 = m0 + 16;
      float v0 = acc0[t][r] + bv;
      float v1 = acc1[t][r] + bv;
      if (RELU) {
        v0 = v0 > 0.f ? v0 : 0.f;
        v1 = v1 > 0.f ? v1 : 0.f;
      }
      if (OUT16) {
        Ch[(long)m0 * N + n] = (_Float16)v0;
        Ch[(long)m1 * N + n] = (_Float16)v1;
      } else {
        Cf[(long)m0 * N + n] = v0;
        Cf[(long)m1 * N + n] = v1;
      }
    }
  }
}

// ---------------------------------------------------------------- normalize
__global__ void k_norm(const float* __restrict__ gin, float* __restrict__ out) {
  __shared__ float red[256];
  int g = blockIdx.x;
  const float* row = gin + (long)g * ODIM;
  float s = 0.f;
  for (int c = threadIdx.x; c < ODIM; c += 256) { float v = row[c]; s += v * v; }
  red[threadIdx.x] = s;
  __syncthreads();
  for (int off = 128; off > 0; off >>= 1) {
    if (threadIdx.x < off) red[threadIdx.x] += red[threadIdx.x + off];
    __syncthreads();
  }
  float norm = sqrtf(red[0]);
  norm = norm > 1e-12f ? norm : 1e-12f;
  float inv = 1.f / norm;
  for (int c = threadIdx.x; c < ODIM; c += 256)
    out[(long)g * ODIM + c] = row[c] * inv;
}

// ---------------------------------------------------------------- host
static inline int cdiv(long a, long b) { return (int)((a + b - 1) / b); }

extern "C" void kernel_launch(void* const* d_in, const int* in_sizes, int n_in,
                              void* d_out, int out_size, void* d_ws, size_t ws_size,
                              hipStream_t stream) {
  const int*   x_cat     = (const int*)d_in[0];
  const int*   edge_attr = (const int*)d_in[1];
  const int*   edge_idx  = (const int*)d_in[2];
  const int*   batch     = (const int*)d_in[3];
  const float* node_emb  = (const float*)d_in[4];
  const float* edge_emb  = (const float*)d_in[5];
  const float* W1  = (const float*)d_in[6];
  const float* b1  = (const float*)d_in[7];
  const float* W2  = (const float*)d_in[8];
  const float* b2  = (const float*)d_in[9];
  const float* pW1 = (const float*)d_in[10];
  const float* pb1 = (const float*)d_in[11];
  const float* pW2 = (const float*)d_in[12];
  const float* pb2 = (const float*)d_in[13];

  char* ws = (char*)d_ws;
  size_t off = 0;
  auto take = [&](size_t bytes) { char* p = ws + off; off += (bytes + 255) & ~(size_t)255; return p; };
  float*    x    = (float*)   take((size_t)NN * DIM * 4);
  float*    h    = (float*)   take((size_t)NN * DIM * 4);
  _Float16* e16  = (_Float16*)take((size_t)NE * DIM * 2);
  _Float16* h16  = (_Float16*)take((size_t)NN * DIM * 2);
  _Float16* t16  = (_Float16*)take((size_t)NN * HID * 2);
  float*    g    = (float*)   take((size_t)NG * DIM * 4);
  _Float16* g16  = (_Float16*)take((size_t)NG * DIM * 2);
  _Float16* gt16 = (_Float16*)take((size_t)NG * DIM * 2);
  float*    graw = (float*)   take((size_t)NG * ODIM * 4);
  _Float16* w1p  = (_Float16*)take((size_t)NLAY * DIM * HID * 2);
  _Float16* w2p  = (_Float16*)take((size_t)NLAY * HID * DIM * 2);
  _Float16* pw1p = (_Float16*)take((size_t)DIM * DIM * 2);
  _Float16* pw2p = (_Float16*)take((size_t)DIM * ODIM * 2);

  // encode
  k_encode_nodes<<<NN, 256, 0, stream>>>(x_cat, node_emb, x);
  k_encode_edges<<<NE, 256, 0, stream>>>(edge_attr, edge_emb, e16);

  // weight repack (f16, WMMA B-fragment order)
  for (int l = 0; l < NLAY; ++l) {
    k_repack_w<<<cdiv((long)DIM * HID, 256), 256, 0, stream>>>(
        W1 + (long)l * DIM * HID, w1p + (long)l * DIM * HID, DIM, HID);
    k_repack_w<<<cdiv((long)HID * DIM, 256), 256, 0, stream>>>(
        W2 + (long)l * HID * DIM, w2p + (long)l * HID * DIM, HID, DIM);
  }
  k_repack_w<<<cdiv((long)DIM * DIM, 256), 256, 0, stream>>>(pW1, pw1p, DIM, DIM);
  k_repack_w<<<cdiv((long)DIM * ODIM, 256), 256, 0, stream>>>(pW2, pw2p, DIM, ODIM);

  // GINE layers
  for (int l = 0; l < NLAY; ++l) {
    k_copy_f32<<<cdiv((long)NN * DIM, 256), 256, 0, stream>>>(x, h, (long)NN * DIM);
    k_edge_scatter<<<cdiv((long)NE * 64, 256), 256, 0, stream>>>(x, e16, edge_idx, h);
    k_f32_to_f16<<<cdiv((long)NN * DIM, 256), 256, 0, stream>>>(h, h16, (long)NN * DIM);
    k_gemm<true, true><<<dim3(cdiv(NN, 256), HID / 64), 256, 0, stream>>>(
        h16, w1p + (long)l * DIM * HID, b1 + (long)l * HID, nullptr, t16, NN, HID, DIM);
    k_gemm<true, false><<<dim3(cdiv(NN, 256), DIM / 64), 256, 0, stream>>>(
        t16, w2p + (long)l * HID * DIM, b2 + (long)l * DIM, x, nullptr, NN, DIM, HID);
  }

  // pool + head MLP + normalize
  k_zero_f32<<<cdiv((long)NG * DIM, 256), 256, 0, stream>>>(g, (long)NG * DIM);
  k_pool<<<cdiv((long)NN * DIM, 256), 256, 0, stream>>>(x, batch, g);
  k_f32_to_f16<<<cdiv((long)NG * DIM, 256), 256, 0, stream>>>(g, g16, (long)NG * DIM);
  k_gemm<true, true><<<dim3(cdiv(NG, 256), DIM / 64), 256, 0, stream>>>(
      g16, pw1p, pb1, nullptr, gt16, NG, DIM, DIM);
  k_gemm<false, false><<<dim3(cdiv(NG, 256), ODIM / 64), 256, 0, stream>>>(
      gt16, pw2p, pb2, graw, nullptr, NG, ODIM, DIM);
  k_norm<<<NG, 256, 0, stream>>>(graw, (float*)d_out);

  (void)in_sizes; (void)n_in; (void)out_size; (void)ws_size;
}
